// Attention_64939905515672
// MI455X (gfx1250) — compile-verified, matrix operand order checked by reading, not run on hardware
//
#include <hip/hip_runtime.h>
#include <hip/hip_bf16.h>
#include <math.h>

typedef __attribute__((ext_vector_type(16))) _Float16 v16h;
typedef __attribute__((ext_vector_type(8)))  _Float16 v8h;
typedef __attribute__((ext_vector_type(8)))  float    v8f;
typedef int aint4 __attribute__((vector_size(16)));   // matches async-LDS builtin

#define DIM     4096
#define NHEADS  32
#define HDIM    128
#define BSZ     4
#define SEQ     512
#define MAXSEQ  2048
#define MROWS   (BSZ * SEQ)   // 2048

// Async global->LDS (CDNA5 ASYNCcnt path), guarded so absence can't break build.
#if defined(__has_builtin)
#if __has_builtin(__builtin_amdgcn_global_load_async_to_lds_b128) && \
    __has_builtin(__builtin_amdgcn_s_wait_asynccnt)
#define HAVE_ASYNC_LDS 1
typedef __attribute__((address_space(1))) char gas_char;
typedef __attribute__((address_space(3))) char las_char;
typedef __attribute__((address_space(1))) aint4 gas_int4;
typedef __attribute__((address_space(3))) aint4 las_int4;
#endif
#endif

// ---------------------------------------------------------------------------
// WMMA fragment loaders (CDNA5 wave32 16x16x32 f16 layouts, CDNA5 ISA)
// A (16x32, MxK): lanes 0-15 -> M=lane, K base +0; lanes 16-31 -> M=lane-16, +8.
// halves [0..7] = K {kb..kb+7}, halves [8..15] = K {kb+16..kb+23}.
// B (32x16, KxN) mirrors A with N in place of M, so the same loader serves a
// row-major [N][K] source (weights W[o][d], K-cache rows [key][d]).
// ---------------------------------------------------------------------------
__device__ __forceinline__ v16h load_frag_rowmajor(const _Float16* __restrict__ P,
                                                   int ld, int r0, int k0, int lane) {
    const int row = r0 + (lane & 15);
    const int kb  = k0 + ((lane >> 4) << 3);
    const _Float16* p = P + (size_t)row * ld + kb;
    v16h f;
#pragma unroll
    for (int i = 0; i < 8; ++i) { f[i] = p[i]; f[8 + i] = p[16 + i]; }
    return f;
}

__device__ __forceinline__ v8f zero8() {
    v8f z = {0.f, 0.f, 0.f, 0.f, 0.f, 0.f, 0.f, 0.f};
    return z;
}

// ---------------------------------------------------------------------------
// x (f32) -> f16 workspace copy
// ---------------------------------------------------------------------------
__global__ void cvt_f32_f16(const float* __restrict__ x, _Float16* __restrict__ y, int n) {
    int i = blockIdx.x * blockDim.x + threadIdx.x;
    if (i < n) y[i] = (_Float16)x[i];
}

// ---------------------------------------------------------------------------
// GEMM: out[m][n] = scale[n] * sum_k A[m][k] * W[n][k]
//   A: f16 row-major MROWS x DIM.  W: int32 (int8 values) row-major DIM x DIM.
// Block: 256 thr = 8 waves. Block tile 256(M) x 64(N). Each wave: 32(M) x 64(N)
// -> 8 WMMAs per K-step against one shared 64x32 LDS weight tile.
// Software pipeline: next int32 tile is loaded into registers while the
// current tile's WMMAs run, so global-load latency hides under the math.
// mode 0/1/2: scatter f16 to (b,h,s,d) q/k/v buffers; mode 2 also writes
// cache_v (f32, (b,pos,h,d)). mode 3: dense f32 row-major store (final out).
// ---------------------------------------------------------------------------
__global__ __launch_bounds__(256) void gemm_w8(const _Float16* __restrict__ A,
                                               const int* __restrict__ W,
                                               const float* __restrict__ S,
                                               int mode,
                                               _Float16* __restrict__ outH,
                                               float* __restrict__ outF) {
    __shared__ _Float16 Wbuf[64 * 32];   // N=64 x K=32 f16 tile (4 KB)
    const int lane  = threadIdx.x & 31;
    const int warp  = threadIdx.x >> 5;
    const int nBase = blockIdx.x * 64;
    const int mBase = blockIdx.y * 256 + warp * 32;

    v8f acc[2][4];
#pragma unroll
    for (int mi = 0; mi < 2; ++mi)
#pragma unroll
        for (int j = 0; j < 4; ++j) acc[mi][j] = zero8();

    // cooperative tile footprint: 2048 ints / 256 thr = 8 contiguous per thread
    const int l   = threadIdx.x * 8;
    const int row = l >> 5;              // 0..63
    const int col = l & 31;              // multiple of 8, no row crossing
    const int* gp = W + (size_t)(nBase + row) * DIM + col;

    int wreg[8];
#pragma unroll
    for (int i = 0; i < 8; ++i) wreg[i] = gp[i];          // tile k0=0

    for (int k0 = 0; k0 < DIM; k0 += 32) {
        __syncthreads();                                   // Wbuf readers done
#pragma unroll
        for (int i = 0; i < 8; ++i) Wbuf[row * 32 + col + i] = (_Float16)wreg[i];
        __syncthreads();

        if (k0 + 32 < DIM) {                               // prefetch next tile
            const int* gn = gp + k0 + 32;
#pragma unroll
            for (int i = 0; i < 8; ++i) wreg[i] = gn[i];
            __builtin_prefetch(gp + k0 + 64, 0, 3);        // global_prefetch_b8
        }

        const v16h a0 = load_frag_rowmajor(A, DIM, mBase,      k0, lane);
        const v16h a1 = load_frag_rowmajor(A, DIM, mBase + 16, k0, lane);
#pragma unroll
        for (int j = 0; j < 4; ++j) {
            const v16h b = load_frag_rowmajor(Wbuf, 32, j * 16, 0, lane);
            acc[0][j] = __builtin_amdgcn_wmma_f32_16x16x32_f16(
                false, a0, false, b, (short)0, acc[0][j], false, false);
            acc[1][j] = __builtin_amdgcn_wmma_f32_16x16x32_f16(
                false, a1, false, b, (short)0, acc[1][j], false, false);
        }
    }

    const int g  = lane >> 4;
    const int nl = lane & 15;
#pragma unroll
    for (int mi = 0; mi < 2; ++mi) {
#pragma unroll
        for (int j = 0; j < 4; ++j) {
            const int   n   = nBase + j * 16 + nl;
            const float scl = S[n];
#pragma unroll
            for (int r = 0; r < 8; ++r) {
                const int   m = mBase + mi * 16 + r + (g << 3);
                const float v = acc[mi][j][r] * scl;
                if (mode == 3) {
                    outF[(size_t)m * DIM + n] = v;
                } else {
                    const int bb = m >> 9, s = m & 511;
                    const int hh = n >> 7, d = n & 127;
                    outH[(((size_t)(bb * NHEADS + hh)) * SEQ + s) * HDIM + d] = (_Float16)v;
                    if (mode == 2)
                        outF[(((size_t)(bb * MAXSEQ + s)) * NHEADS + hh) * HDIM + d] = v;
                }
            }
        }
    }
}

// ---------------------------------------------------------------------------
// RoPE on qh/kh (f16, (b,h,s,d)); for K also writes cache_k (f32, (b,pos,h,d)).
// grid.y: 0 = Q, 1 = K.
// ---------------------------------------------------------------------------
__global__ void rope_cache(_Float16* __restrict__ qh, _Float16* __restrict__ kh,
                           const float* __restrict__ fc, const float* __restrict__ fs,
                           float* __restrict__ cache_k) {
    const int isK = blockIdx.y;
    int t = blockIdx.x * blockDim.x + threadIdx.x;     // over BSZ*NHEADS*SEQ*64
    const int i = t & 63;  t >>= 6;
    const int s = t & 511; t >>= 9;
    const int h = t & 31;  t >>= 5;
    const int b = t;
    _Float16* p = (isK ? kh : qh) +
                  (((size_t)(b * NHEADS + h)) * SEQ + s) * HDIM + 2 * i;
    const float c  = fc[s * 64 + i];
    const float sn = fs[s * 64 + i];
    const float tr = (float)p[0], ti = (float)p[1];
    const float orr = tr * c - ti * sn;
    const float oii = tr * sn + ti * c;
    p[0] = (_Float16)orr;
    p[1] = (_Float16)oii;
    if (isK) {
        float* ck = cache_k + (((size_t)(b * MAXSEQ + s)) * NHEADS + h) * HDIM + 2 * i;
        ck[0] = orr;
        ck[1] = oii;
    }
}

// ---------------------------------------------------------------------------
// Flash attention. One wave handles (b, h, 16-row q tile). Key blocks of 32.
// Causal mask means keys >= 512 never contribute (they are masked anyway).
// S = Q*K^T via 8 WMMAs / block; online softmax with shfl_xor row reductions
// (rows live in 16-lane half-groups of the C layout); P goes through LDS to
// reach the A-fragment layout; V block staged in LDS (async DMA if available)
// for the strided B-fragment reads.
// ---------------------------------------------------------------------------
__global__ __launch_bounds__(128) void attn(const _Float16* __restrict__ qh,
                                            const _Float16* __restrict__ kh,
                                            const _Float16* __restrict__ vh,
                                            _Float16* __restrict__ ctx) {
    __shared__ _Float16 Pbuf[4][16 * 32];     // per-wave P tile (A-frag source)
    __shared__ _Float16 Vbuf[4][32 * HDIM];   // per-wave V block (8 KB each)
    const int lane = threadIdx.x & 31;
    const int warp = threadIdx.x >> 5;
    int gw = blockIdx.x * 4 + warp;
    const int qt = gw & 31; gw >>= 5;
    const int h  = gw & 31; gw >>= 5;
    const int b  = gw;

    const _Float16* Q = qh + ((size_t)(b * NHEADS + h)) * SEQ * HDIM;
    const _Float16* K = kh + ((size_t)(b * NHEADS + h)) * SEQ * HDIM;
    const _Float16* V = vh + ((size_t)(b * NHEADS + h)) * SEQ * HDIM;

    v16h qf[4];                                // Q tile 16x128 resident
#pragma unroll
    for (int t = 0; t < 4; ++t) qf[t] = load_frag_rowmajor(Q, HDIM, qt * 16, t * 32, lane);

    v8f o[8];
#pragma unroll
    for (int j = 0; j < 8; ++j) o[j] = zero8();
    float rmax[8], rsum[8];
#pragma unroll
    for (int r = 0; r < 8; ++r) { rmax[r] = -INFINITY; rsum[r] = 0.f; }

    const int   g     = lane >> 4;
    const int   nl    = lane & 15;
    const float scale = 0.08838834764831845f;  // 1/sqrt(128)
    const int   q0    = qt * 16;
    const int   nb    = (qt + 2) >> 1;         // causal #key-blocks of 32

    for (int kb = 0; kb < nb; ++kb) {
        // stage V block (32 keys x 128 = contiguous 8 KB) into LDS
#if defined(HAVE_ASYNC_LDS)
        {
            gas_char* gpc = (gas_char*)(char*)(V + (size_t)(kb * 32) * HDIM);
            las_char* lpc = (las_char*)(char*)&Vbuf[warp][0];
            gpc += lane * 16;
            lpc += lane * 16;
#pragma unroll
            for (int i = 0; i < 16; ++i)
                __builtin_amdgcn_global_load_async_to_lds_b128(
                    (gas_int4*)(gpc + i * 512), (las_int4*)(lpc + i * 512), 0, 0);
            __builtin_amdgcn_s_wait_asynccnt(0);
        }
#else
        {
            const v8h* src = (const v8h*)(V + (size_t)(kb * 32 + lane) * HDIM);
            v8h*       dst = (v8h*)&Vbuf[warp][lane * HDIM];
#pragma unroll
            for (int i = 0; i < 16; ++i) dst[i] = src[i];
        }
#endif

        v8f s0 = zero8(), s1 = zero8();        // keys [0..15], [16..31] of block
#pragma unroll
        for (int t = 0; t < 4; ++t) {
            const v16h b0 = load_frag_rowmajor(K, HDIM, kb * 32,      t * 32, lane);
            const v16h b1 = load_frag_rowmajor(K, HDIM, kb * 32 + 16, t * 32, lane);
            s0 = __builtin_amdgcn_wmma_f32_16x16x32_f16(false, qf[t], false, b0, (short)0, s0, false, false);
            s1 = __builtin_amdgcn_wmma_f32_16x16x32_f16(false, qf[t], false, b1, (short)0, s1, false, false);
        }

        float nm[8];
#pragma unroll
        for (int r = 0; r < 8; ++r) {
            const int qrow = q0 + r + (g << 3);
            const int key0 = kb * 32 + nl;
            const int key1 = key0 + 16;
            float a0 = s0[r] * scale + (key0 <= qrow ? 0.f : -1e9f);
            float a1 = s1[r] * scale + (key1 <= qrow ? 0.f : -1e9f);
            s0[r] = a0; s1[r] = a1;
            float mx = fmaxf(a0, a1);
            mx = fmaxf(mx, __shfl_xor(mx, 1));
            mx = fmaxf(mx, __shfl_xor(mx, 2));
            mx = fmaxf(mx, __shfl_xor(mx, 4));
            mx = fmaxf(mx, __shfl_xor(mx, 8));
            nm[r] = fmaxf(rmax[r], mx);
        }
#pragma unroll
        for (int r = 0; r < 8; ++r) {
            const float corr = __expf(rmax[r] - nm[r]);
            rmax[r] = nm[r];
            const float p0 = __expf(s0[r] - nm[r]);
            const float p1 = __expf(s1[r] - nm[r]);
            float ps = p0 + p1;
            ps += __shfl_xor(ps, 1);
            ps += __shfl_xor(ps, 2);
            ps += __shfl_xor(ps, 4);
            ps += __shfl_xor(ps, 8);
            rsum[r] = rsum[r] * corr + ps;
            s0[r] = p0; s1[r] = p1;
#pragma unroll
            for (int j = 0; j < 8; ++j) o[j][r] *= corr;
        }

        {   // P (16x32) -> LDS in row-major so it can be read as an A fragment
            _Float16* pb = Pbuf[warp];
#pragma unroll
            for (int r = 0; r < 8; ++r) {
                const int prow = r + (g << 3);
                pb[prow * 32 + nl]      = (_Float16)s0[r];
                pb[prow * 32 + 16 + nl] = (_Float16)s1[r];
            }
        }
        v16h pf;
        {
            const _Float16* pb = Pbuf[warp] + (size_t)(lane & 15) * 32 + ((lane >> 4) << 3);
#pragma unroll
            for (int i = 0; i < 8; ++i) { pf[i] = pb[i]; pf[8 + i] = pb[16 + i]; }
        }
#pragma unroll
        for (int j = 0; j < 8; ++j) {           // O(16x128) += P(16x32) * V(32x128)
            const _Float16* vb = Vbuf[warp];
            const int d     = j * 16 + nl;
            const int kbase = g << 3;
            v16h bf;
#pragma unroll
            for (int i = 0; i < 8; ++i) {
                bf[i]     = vb[(kbase + i) * HDIM + d];
                bf[8 + i] = vb[(kbase + 16 + i) * HDIM + d];
            }
            o[j] = __builtin_amdgcn_wmma_f32_16x16x32_f16(false, pf, false, bf, (short)0, o[j], false, false);
        }
    }

#pragma unroll
    for (int r = 0; r < 8; ++r) {
        const float inv = 1.0f / rsum[r];
        const int   m   = b * SEQ + q0 + r + (g << 3);
#pragma unroll
        for (int j = 0; j < 8; ++j)
            ctx[(size_t)m * DIM + h * HDIM + j * 16 + nl] = (_Float16)(o[j][r] * inv);
    }
}

// ---------------------------------------------------------------------------
// Host-side orchestration (graph-capture safe: only async stream ops).
// Inputs: 0:x 1:cos 2:sin 3:mask 4:idx 5:cache_k 6:cache_v
//         7:wq_q 8:wq_s 9:wk_q 10:wk_s 11:wv_q 12:wv_s 13:wo_q 14:wo_s
// d_out: [out (2048x4096) | cache_k (4x2048x32x128) | cache_v (same)] f32
// ---------------------------------------------------------------------------
extern "C" void kernel_launch(void* const* d_in, const int* in_sizes, int n_in,
                              void* d_out, int out_size, void* d_ws, size_t ws_size,
                              hipStream_t stream) {
    (void)in_sizes; (void)n_in; (void)out_size; (void)ws_size;
    const float* x  = (const float*)d_in[0];
    const float* fc = (const float*)d_in[1];
    const float* fs = (const float*)d_in[2];
    const int*   wq = (const int*)d_in[7];  const float* sq = (const float*)d_in[8];
    const int*   wk = (const int*)d_in[9];  const float* sk = (const float*)d_in[10];
    const int*   wv = (const int*)d_in[11]; const float* sv = (const float*)d_in[12];
    const int*   wo = (const int*)d_in[13]; const float* so = (const float*)d_in[14];

    float* out    = (float*)d_out;
    float* cacheK = out + (size_t)MROWS * DIM;
    float* cacheV = cacheK + (size_t)BSZ * MAXSEQ * NHEADS * HDIM;

    const size_t actElems = (size_t)MROWS * DIM;      // 8,388,608 f16 each
    _Float16* xh  = (_Float16*)d_ws;
    _Float16* qh  = xh + actElems;
    _Float16* kh  = qh + actElems;
    _Float16* vh  = kh + actElems;
    _Float16* ctx = vh + actElems;                    // total ws: ~84 MB

    // zero both caches (positions >= SEQ must read back as 0)
    (void)hipMemsetAsync(cacheK, 0,
                         (size_t)2 * BSZ * MAXSEQ * NHEADS * HDIM * sizeof(float), stream);

    cvt_f32_f16<<<(MROWS * DIM) / 256, 256, 0, stream>>>(x, xh, MROWS * DIM);

    dim3 gg(DIM / 64, MROWS / 256);
    gemm_w8<<<gg, 256, 0, stream>>>(xh, wq, sq, 0, qh, nullptr);
    gemm_w8<<<gg, 256, 0, stream>>>(xh, wk, sk, 1, kh, nullptr);
    gemm_w8<<<gg, 256, 0, stream>>>(xh, wv, sv, 2, vh, cacheV);

    rope_cache<<<dim3((BSZ * NHEADS * SEQ * 64) / 256, 2), 256, 0, stream>>>(
        qh, kh, fc, fs, cacheK);

    attn<<<(BSZ * NHEADS * (SEQ / 16)) / 4, 128, 0, stream>>>(qh, kh, vh, ctx);

    gemm_w8<<<gg, 256, 0, stream>>>(ctx, wo, so, 3, nullptr, out);
}
